// CKConv_10368051052953
// MI455X (gfx1250) — matrix operand order repaired and checked by reading, not compiled, not run on hardware
//
#include <hip/hip_runtime.h>

// ---- problem constants (from reference) ----
#define OMEGA0 30.0f
#define CIN    16
#define COUT   32
#define HID    32
#define BATCH  8
#define LSEQ   2048
#define NCOPY  4           // staggered reversed-x copies (8B-aligned B loads)
#define XR_LEN 2112        // halves per reversed row (2048 data + slack), 16B mult
#define LDS_W  2112        // max staged halves per copy

typedef __attribute__((ext_vector_type(16))) _Float16 v16h;
typedef __attribute__((ext_vector_type(8)))  float    v8f;
typedef __attribute__((ext_vector_type(4)))  unsigned int v4u;
typedef __attribute__((ext_vector_type(8)))  int      v8i;
typedef __attribute__((ext_vector_type(4)))  int      v4i;

// ---------------------------------------------------------------------------
// Kernel 1: SIREN MLP over positions; writes time-reversed f16 conv weights
//   gh[(o*CIN+i)*LSEQ + d] = k[o*CIN+i, 2047-d]   (lag-major, contiguous in d)
// ---------------------------------------------------------------------------
__global__ __launch_bounds__(128) void siren_gen(
    const float* __restrict__ w1, const float* __restrict__ b1,
    const float* __restrict__ w2, const float* __restrict__ b2,
    const float* __restrict__ w3, const float* __restrict__ b3,
    _Float16* __restrict__ gh)
{
    const int l = blockIdx.x * blockDim.x + threadIdx.x;
    if (l >= LSEQ) return;
    const float t = -1.0f + 2.0f * (float)l / (float)(LSEQ - 1);

    float h1[HID], h2[HID];
#pragma unroll
    for (int j = 0; j < HID; ++j)
        h1[j] = __sinf(OMEGA0 * (w1[j] * t + b1[j]));
#pragma unroll 4
    for (int j = 0; j < HID; ++j) {
        float s = b2[j];
#pragma unroll
        for (int k = 0; k < HID; ++k) s += w2[j * HID + k] * h1[k];
        h2[j] = __sinf(OMEGA0 * s);
    }
    const int drev = (LSEQ - 1) - l;        // time reversal: lag d = 2047 - l
    for (int r = 0; r < COUT * CIN; ++r) {
        float s = b3[r];
#pragma unroll
        for (int k = 0; k < HID; ++k) s += w3[r * HID + k] * h2[k];
        gh[(size_t)r * LSEQ + drev] = (_Float16)s;
    }
}

// ---------------------------------------------------------------------------
// Kernel 2: build 4 staggered time-reversed f16 copies of x:
//   xr[s][b*CIN+i][u] = (u+s < 2048) ? x[b,i, 2047-(u+s)] : 0
// ---------------------------------------------------------------------------
__global__ __launch_bounds__(256) void xrev_fill(
    const float* __restrict__ x, _Float16* __restrict__ xr)
{
    const int idx = blockIdx.x * blockDim.x + threadIdx.x;
    if (idx >= NCOPY * BATCH * CIN * XR_LEN) return;
    const int u   = idx % XR_LEN;
    const int r   = idx / XR_LEN;
    const int row = r % (BATCH * CIN);
    const int s   = r / (BATCH * CIN);
    const int w   = u + s;
    float v = 0.0f;
    if (w < LSEQ) v = x[(size_t)row * LSEQ + (LSEQ - 1 - w)];
    xr[idx] = (_Float16)v;
}

// ---------------------------------------------------------------------------
// Kernel 3: causal long conv as tiled f16 WMMA GEMM (f32 accumulate).
// One wave32 per 16-position tile; all 32 output channels per wave
// (2 accumulators sharing one B fragment). TDM double-buffers the x window
// into LDS (memory->LDS DMA, TENSORcnt) while WMMAs run.
// ---------------------------------------------------------------------------
__global__ __launch_bounds__(32) void conv_wmma(
    const _Float16* __restrict__ gh,      // [COUT*CIN][LSEQ] time-reversed
    const _Float16* __restrict__ xr,      // [NCOPY][B*CIN][XR_LEN] reversed
    const float*    __restrict__ conv_bias,
    float*          __restrict__ out)     // [B][COUT][LSEQ]
{
    __shared__ __align__(16) _Float16 ldsb[2][NCOPY * LDS_W];

    const int lane = threadIdx.x;           // 0..31
    const int lt   = blockIdx.x;            // 0..127 position tile
    const int b    = blockIdx.y;            // 0..7   batch
    const int l0   = lt * 16;

    // A-fragment lane roles (ISA 16-bit 16x32 A layout)
    const int mrow = lane & 15;
    const int kA   = (lane < 16) ? 0 : 8;
    // B-fragment lane roles (32x16 B: lane = column, half-wave = K halves)
    const int col  = lane & 15;
    const int kB   = (lane < 16) ? 0 : 8;
    const int Cb   = kB + 15 - col;         // per-lane window offset (l0-indep)
    const int sel  = Cb & (NCOPY - 1);      // staggered copy giving 8B align

    const int nchunks = (l0 + 15) / 32 + 1; // lag chunks of 32 (causal bound)
    const int size    = l0 + 64;            // staged halves per copy (16-mult)
    const int ubase   = 2032 - l0;          // 16-half-aligned xr offset

    // ---- TDM descriptor pieces (wave-uniform) ----
    const unsigned long long copy_stride = (unsigned long long)BATCH * CIN * XR_LEN; // elems
    const unsigned int ldsaddr[2] = {
        (unsigned int)(uintptr_t)&ldsb[0][0],
        (unsigned int)(uintptr_t)&ldsb[1][0] };

    v8i g1;
    g1[0] = 0x00010000;                       // data_size = 1 (2 bytes)
    g1[1] = (size & 0xffff) << 16;            // tensor_dim0 lo
    g1[2] = (NCOPY << 16);                    // tensor_dim0 hi=0 | tensor_dim1=4
    g1[3] = (size & 0xffff) << 16;            // tensor_dim1 hi=0 | tile_dim0
    g1[4] = NCOPY;                            // tile_dim1=4, tile_dim2=0
    g1[5] = (int)(copy_stride & 0xffffffffu); // tensor_dim0_stride lo
    g1[6] = (int)(copy_stride >> 32);         // stride hi | dim1_stride lo = 0
    g1[7] = 0;
    const v4i gz4 = {0, 0, 0, 0};
    const v8i gz8 = {0, 0, 0, 0, 0, 0, 0, 0};

    auto issue_tdm = [&](int i, int buf) {
        const unsigned long long ga = (unsigned long long)(uintptr_t)
            (xr + ((size_t)(b * CIN + i)) * XR_LEN + ubase);
        v4u g0;
        g0.x = 0x1u;                               // count=1, load descriptor
        g0.y = ldsaddr[buf];                       // LDS byte offset
        g0.z = (unsigned int)ga;                   // global_addr[31:0]
        g0.w = (unsigned int)(ga >> 32) | 0x80000000u; // ga[56:32] | type=2
        __builtin_amdgcn_tensor_load_to_lds(g0, g1, gz4, gz4, gz8, 0);
    };

    v8f acc0 = {}, acc1 = {};

    issue_tdm(0, 0);
    for (int i = 0; i < CIN; ++i) {
        const int buf = i & 1;
        if (i + 1 < CIN) {
            issue_tdm(i + 1, buf ^ 1);
            __builtin_amdgcn_s_wait_tensorcnt(1);  // buffer `buf` is ready
        } else {
            __builtin_amdgcn_s_wait_tensorcnt(0);
        }
        asm volatile("" ::: "memory");             // order LDS reads after DMA

        const _Float16* grow0 = gh + ((size_t)(mrow * CIN + i)) * LSEQ;
        const _Float16* grow1 = gh + ((size_t)((16 + mrow) * CIN + i)) * LSEQ;
        // 8B-aligned base pointer: copy `sel` starts at halfword sel*size
        const uint2* bbase =
            reinterpret_cast<const uint2*>(&ldsb[buf][sel * size + (Cb - sel)]);

        for (int c = 0; c < nchunks; ++c) {
            const int d0 = c * 32;

            // B: two runs of 8 halves (k 8h..8h+7, 16+8h..16+8h+7), 4x ds_load_b64
            v16h bm;
            const uint2* p = bbase + c * 8;   // 32 halves/chunk = 8 uint2
            uint2* bmp = reinterpret_cast<uint2*>(&bm);
            bmp[0] = p[0]; bmp[1] = p[1]; bmp[2] = p[4]; bmp[3] = p[5];

            // A: weights for both M-tiles, aligned 16B global loads
            v16h a0, a1;
            *reinterpret_cast<float4*>(&a0) =
                *reinterpret_cast<const float4*>(&grow0[d0 + kA]);
            *(reinterpret_cast<float4*>(&a0) + 1) =
                *reinterpret_cast<const float4*>(&grow0[d0 + 16 + kA]);
            *reinterpret_cast<float4*>(&a1) =
                *reinterpret_cast<const float4*>(&grow1[d0 + kA]);
            *(reinterpret_cast<float4*>(&a1) + 1) =
                *reinterpret_cast<const float4*>(&grow1[d0 + 16 + kA]);

            acc0 = __builtin_amdgcn_wmma_f32_16x16x32_f16(
                false, a0, false, bm, (short)0, acc0, false, false);
            acc1 = __builtin_amdgcn_wmma_f32_16x16x32_f16(
                false, a1, false, bm, (short)0, acc1, false, false);
        }
        asm volatile("" ::: "memory");             // reads done before reuse
    }

    // ---- write D: lane = column n, VGPR v -> M = v (+8 for upper half-wave)
    const int moff = (lane < 16) ? 0 : 8;
#pragma unroll
    for (int v = 0; v < 8; ++v) {
        const int r0 = moff + v;
        const int r1 = 16 + moff + v;
        out[((size_t)(b * COUT + r0)) * LSEQ + l0 + col] =
            acc0[v] + conv_bias[r0];
        out[((size_t)(b * COUT + r1)) * LSEQ + l0 + col] =
            acc1[v] + conv_bias[r1];
    }
}

// ---------------------------------------------------------------------------
extern "C" void kernel_launch(void* const* d_in, const int* in_sizes, int n_in,
                              void* d_out, int out_size, void* d_ws, size_t ws_size,
                              hipStream_t stream)
{
    const float* x         = (const float*)d_in[0];
    const float* w1        = (const float*)d_in[1];
    const float* b1        = (const float*)d_in[2];
    const float* w2        = (const float*)d_in[3];
    const float* b2        = (const float*)d_in[4];
    const float* w3        = (const float*)d_in[5];
    const float* b3        = (const float*)d_in[6];
    const float* conv_bias = (const float*)d_in[7];
    float*       out       = (float*)d_out;

    // workspace layout: gh (2 MiB) | xr (4 copies, ~2.1 MiB)
    _Float16* gh = (_Float16*)d_ws;
    _Float16* xr = (_Float16*)((char*)d_ws + (size_t)COUT * CIN * LSEQ * 2);

    siren_gen<<<LSEQ / 128, 128, 0, stream>>>(w1, b1, w2, b2, w3, b3, gh);

    const int nx = NCOPY * BATCH * CIN * XR_LEN;
    xrev_fill<<<(nx + 255) / 256, 256, 0, stream>>>(x, xr);

    conv_wmma<<<dim3(LSEQ / 16, BATCH), 32, 0, stream>>>(
        gh, xr, conv_bias, out);
}
